// BIGRU_68298569941081
// MI455X (gfx1250) — matrix operand order repaired and checked by reading, not compiled, not run on hardware
//
#include <hip/hip_runtime.h>
#include <hip/hip_bf16.h>

// ---------------------------------------------------------------------------
// BiGRU for MI455X (gfx1250): f16 WMMA GEMMs (v_wmma_f32_16x16x32_f16) with
// f32 accumulation; async global->LDS double-buffered staging
// (GLOBAL_LOAD_ASYNC_TO_LDS_B128 + s_wait_asynccnt) feeding register-
// double-buffered ds_load_b128 so WMMAs overlap LDS latency (partial
// s_wait_dscnt); fused GRU gate pointwise.
// NOTE: reference applies _gru_dir identically for d=0,1 (both forward in
// time, no reversal) -- implemented exactly that way.
// ---------------------------------------------------------------------------

#define H_   1024
#define T_   64
#define N_   256
#define C_   2048      // 2*H (bidir concat)
#define IN_  150
#define INP_ 256       // padded input width (multiple of 128 for K-chunking)
#define NC_  60

// LDS staging: K-chunk = 128 halves (256 B/row), row pitch 272 B (17*16B) so
// 16 rows at the same byte offset cover all 64 banks conflict-free.
#define KCH_      128
#define ROWP_     272
#define AW_BYTES  (16 * ROWP_)       // one wave's 16x128 A tile  (4352 B)
#define A_BYTES   (4 * AW_BYTES)     // 4 waves                  (17408 B)
#define BG_BYTES  (16 * ROWP_)       // one gate's 16x128 B tile  (4352 B)
#define B_BYTES   (3 * BG_BYTES)     // 3 gates                  (13056 B)
#define BUF_BYTES (A_BYTES + B_BYTES)

typedef _Float16 h8  __attribute__((ext_vector_type(8)));
typedef _Float16 h16 __attribute__((ext_vector_type(16)));
typedef float    f8  __attribute__((ext_vector_type(8)));

// ---- CDNA5 async global->LDS copy (ASYNCcnt-tracked) ----------------------
__device__ __forceinline__ void async_b128(unsigned lds_off, const void* gaddr) {
  asm volatile("global_load_async_to_lds_b128 %0, %1, off"
               :: "v"(lds_off), "v"(gaddr)
               : "memory");
}
__device__ __forceinline__ void wait_async0() {
  asm volatile("s_wait_asynccnt 0x0" ::: "memory");
}

__device__ __forceinline__ h16 cat8(h8 lo, h8 hi) {
  return __builtin_shufflevector(lo, hi, 0,1,2,3,4,5,6,7,8,9,10,11,12,13,14,15);
}

__device__ __forceinline__ float sigmoidf_(float x) {
  return 1.0f / (1.0f + __expf(-x));
}

// ---- prefetch one K-chunk (128 halves) of A (per-wave) and B (per-block) --
__device__ __forceinline__ void prefetch_chunk(
    unsigned bufBase,                 // LDS byte offset of ping/pong buffer
    const _Float16* __restrict__ gA,  // A base: row (m0 of this wave), col 0
    int ldA,                          // A row stride (elements)
    const _Float16* __restrict__ gB,  // W (this dir): rows 3H x ldB
    int ldB, int j0, int k0,
    int wave, int lane, int tid) {
  // A: wave-private 16x128 tile, 8 b128 issues per wave (32 lanes x 16B each)
  unsigned aBase = bufBase + wave * AW_BYTES;
#pragma unroll
  for (int i = 0; i < 8; ++i) {
    int seg = lane + 32 * i;          // 0..255 -> 16 rows x 16 segs
    int row = seg >> 4;
    int byt = (seg & 15) * 16;
    async_b128(aBase + row * ROWP_ + byt,
               gA + (size_t)row * ldA + k0 + (byt >> 1));
  }
  // B: block-wide 3 gate tiles of 16x128; 2 issues per gate (128 lanes x 2)
  unsigned bBase = bufBase + A_BYTES;
#pragma unroll
  for (int g = 0; g < 3; ++g) {
#pragma unroll
    for (int i = 0; i < 2; ++i) {
      int seg = tid + 128 * i;        // 0..255
      int row = seg >> 4;
      int byt = (seg & 15) * 16;
      async_b128(bBase + g * BG_BYTES + row * ROWP_ + byt,
                 gB + (size_t)(g * H_ + j0 + row) * ldB + k0 + (byt >> 1));
    }
  }
}

// ---- load one 32-wide K sub-iter's operands from LDS ----------------------
// A layout: lane m (<16): halves 0..15=K+0..15;
// lane m+16: K+16..31. B: lane n holds 16 contiguous K halves,
// lane>=16 takes the upper K half.
__device__ __forceinline__ void load_set(const char* __restrict__ sbuf,
                                         int wave, int lane, int kc,
                                         h16& a, h16& b0, h16& b1, h16& b2) {
  const int m   = lane & 15;
  const int hiL = lane >> 4;
  const char* ap = sbuf + wave * AW_BYTES + m * ROWP_ + kc * 64 + hiL * 32;
  a = cat8(*(const h8*)(ap), *(const h8*)(ap + 16));
  const char* bp = sbuf + A_BYTES + m * ROWP_ + kc * 64 + hiL * 32;
  b0 = cat8(*(const h8*)(bp),                *(const h8*)(bp + 16));
  b1 = cat8(*(const h8*)(bp + BG_BYTES),     *(const h8*)(bp + BG_BYTES + 16));
  b2 = cat8(*(const h8*)(bp + 2 * BG_BYTES), *(const h8*)(bp + 2 * BG_BYTES + 16));
}

// ---- consume one K-chunk: 4 sub-iters x 3 WMMAs, reg double-buffered ------
__device__ __forceinline__ void compute_chunk(
    const char* __restrict__ sbuf, int wave, int lane,
    f8& accR, f8& accZ, f8& acc3) {
  h16 a, b0, b1, b2, an, b0n, b1n, b2n;
  load_set(sbuf, wave, lane, 0, a, b0, b1, b2);
#pragma unroll
  for (int kc = 0; kc < 4; ++kc) {
    if (kc < 3) load_set(sbuf, wave, lane, kc + 1, an, b0n, b1n, b2n);
    accR = __builtin_amdgcn_wmma_f32_16x16x32_f16(false, a, false, b0,
                                                  (short)0, accR, false, false);
    accZ = __builtin_amdgcn_wmma_f32_16x16x32_f16(false, a, false, b1,
                                                  (short)0, accZ, false, false);
    acc3 = __builtin_amdgcn_wmma_f32_16x16x32_f16(false, a, false, b2,
                                                  (short)0, acc3, false, false);
    a = an; b0 = b0n; b1 = b1n; b2 = b2n;
  }
}

// ---- BatchNorm stats ------------------------------------------------------
__global__ void bn_stats_kernel(const float* __restrict__ x,
                                const float* __restrict__ gamma,
                                const float* __restrict__ beta,
                                float* __restrict__ stats) {
  __shared__ float ss[256], sq[256];
  int f = blockIdx.x;
  float s = 0.f, q = 0.f;
  for (int r = threadIdx.x; r < N_ * T_; r += 256) {
    float v = x[(size_t)r * IN_ + f];
    s += v; q += v * v;
  }
  ss[threadIdx.x] = s; sq[threadIdx.x] = q;
  __syncthreads();
  for (int off = 128; off > 0; off >>= 1) {
    if ((int)threadIdx.x < off) {
      ss[threadIdx.x] += ss[threadIdx.x + off];
      sq[threadIdx.x] += sq[threadIdx.x + off];
    }
    __syncthreads();
  }
  if (threadIdx.x == 0) {
    float inv  = 1.0f / (float)(N_ * T_);
    float mean = ss[0] * inv;
    float var  = sq[0] * inv - mean * mean;
    float scale = rsqrtf(var + 1e-5f) * gamma[f];
    stats[f]        = scale;
    stats[INP_ + f] = beta[f] - mean * scale;
  }
}

// ---- Apply BN, convert to f16, pad 150 -> 256 -----------------------------
__global__ void bn_apply_kernel(const float* __restrict__ x,
                                const float* __restrict__ stats,
                                _Float16* __restrict__ xn) {
  int idx = blockIdx.x * 256 + threadIdx.x;
  if (idx >= N_ * T_ * INP_) return;
  int f = idx % INP_;
  int r = idx / INP_;
  float v = 0.f;
  if (f < IN_) v = x[(size_t)r * IN_ + f] * stats[f] + stats[INP_ + f];
  xn[idx] = (_Float16)v;
}

// ---- f32 -> f16 weight convert with K padding -----------------------------
__global__ void cvt_pad_kernel(const float* __restrict__ src,
                               _Float16* __restrict__ dst,
                               int rows, int K, int Kp) {
  long idx = (long)blockIdx.x * 256 + threadIdx.x;
  if (idx >= (long)rows * Kp) return;
  int c = (int)(idx % Kp);
  int r = (int)(idx / Kp);
  float v = (c < K) ? src[(size_t)r * K + c] : 0.f;
  dst[idx] = (_Float16)v;
}

// ---- Fused GRU step -------------------------------------------------------
// grid: (jTiles=64, mBlocks=4, dir=2); block: 128 (4 waves). Each wave owns a
// 16(batch)x16(j) tile for all 3 gates; B tiles shared block-wide via LDS.
__global__ __launch_bounds__(128)
void gru_step_kernel(const _Float16* __restrict__ xin, int Kin,
                     _Float16* __restrict__ hbuf,        // [N][T][2H] f16 out / h_prev
                     const _Float16* __restrict__ wih,   // [2][3H][Kin] f16
                     const _Float16* __restrict__ whh,   // [2][3H][H] f16
                     const float* __restrict__ bih,      // [2][3H]
                     const float* __restrict__ bhh,      // [2][3H]
                     float* __restrict__ enc,            // encoder f32 out or null
                     int t, int first) {
  __shared__ __align__(16) char smem[2 * BUF_BYTES];
  const int tid  = threadIdx.x;
  const int wave = tid >> 5;
  const int lane = tid & 31;
  const int j0   = blockIdx.x * 16;
  const int m0   = blockIdx.y * 64 + wave * 16;
  const int dir  = blockIdx.z;
  const unsigned smem0 = (unsigned)(uintptr_t)smem;   // LDS byte offset

  f8 accR = {}, accZ = {}, accI = {}, accN = {};

  // ---- phase 1: gi_t = x_t @ w_ih^T ----
  {
    const _Float16* gA = xin + (size_t)t * Kin + (size_t)m0 * (T_ * Kin);
    const _Float16* gB = wih + (size_t)dir * 3 * H_ * Kin;
    const int nc = Kin / KCH_;
    prefetch_chunk(smem0, gA, T_ * Kin, gB, Kin, j0, 0, wave, lane, tid);
    for (int c = 0; c < nc; ++c) {
      wait_async0();
      __syncthreads();
      if (c + 1 < nc)
        prefetch_chunk(smem0 + ((c + 1) & 1) * BUF_BYTES, gA, T_ * Kin,
                       gB, Kin, j0, (c + 1) * KCH_, wave, lane, tid);
      compute_chunk(smem + (c & 1) * BUF_BYTES, wave, lane, accR, accZ, accI);
    }
    __syncthreads();
  }

  // ---- phase 2: gh_t = h_{t-1} @ w_hh^T (skip at t==0; h_prev = 0) ----
  if (!first) {
    const _Float16* gA = hbuf + (size_t)(t - 1) * C_ + dir * H_
                              + (size_t)m0 * (T_ * C_);
    const _Float16* gB = whh + (size_t)dir * 3 * H_ * H_;
    const int nc = H_ / KCH_;
    prefetch_chunk(smem0, gA, T_ * C_, gB, H_, j0, 0, wave, lane, tid);
    for (int c = 0; c < nc; ++c) {
      wait_async0();
      __syncthreads();
      if (c + 1 < nc)
        prefetch_chunk(smem0 + ((c + 1) & 1) * BUF_BYTES, gA, T_ * C_,
                       gB, H_, j0, (c + 1) * KCH_, wave, lane, tid);
      compute_chunk(smem + (c & 1) * BUF_BYTES, wave, lane, accR, accZ, accN);
    }
  }

  // ---- fused GRU gates on the C/D register layout ----
  // VGPR i -> M = m0 + i + (lane>=16 ? 8 : 0), N = j0 + (lane&15)
  const float* bihd = bih + dir * 3 * H_;
  const float* bhhd = bhh + dir * 3 * H_;
  const int j = j0 + (lane & 15);
  const float br_  = bihd[j]          + bhhd[j];
  const float bz_  = bihd[H_ + j]     + bhhd[H_ + j];
  const float bin_ = bihd[2 * H_ + j];
  const float bhn_ = bhhd[2 * H_ + j];
  const int moff = m0 + ((lane >> 4) << 3);
#pragma unroll
  for (int i = 0; i < 8; ++i) {
    int nrow = moff + i;
    float r  = sigmoidf_(accR[i] + br_);
    float z  = sigmoidf_(accZ[i] + bz_);
    float nn = tanhf(accI[i] + bin_ + r * (accN[i] + bhn_));
    float hp = first ? 0.0f
                     : (float)hbuf[((size_t)nrow * T_ + (t - 1)) * C_ + dir * H_ + j];
    float h  = (1.0f - z) * nn + z * hp;
    size_t oidx = ((size_t)nrow * T_ + t) * C_ + dir * H_ + j;
    hbuf[oidx] = (_Float16)h;
    if (enc) enc[oidx] = h;
  }
}

// ---- FC head --------------------------------------------------------------
__global__ void fc_kernel(const float* __restrict__ enc,
                          const float* __restrict__ fcw,
                          const float* __restrict__ fcb,
                          float* __restrict__ out) {
  int n = blockIdx.x;
  int k = threadIdx.x;
  if (k >= NC_) return;
  const float* hrow = enc + ((size_t)n * T_ + (T_ - 1)) * C_;
  const float* wr   = fcw + (size_t)k * C_;
  float acc = fcb[k];
  for (int c = 0; c < C_; ++c) acc += hrow[c] * wr[c];
  out[(size_t)n * NC_ + k] = acc;
}

// ---------------------------------------------------------------------------
extern "C" void kernel_launch(void* const* d_in, const int* in_sizes, int n_in,
                              void* d_out, int out_size, void* d_ws, size_t ws_size,
                              hipStream_t stream) {
  (void)in_sizes; (void)n_in; (void)out_size; (void)ws_size;
  const float* x     = (const float*)d_in[0];
  const float* gamma = (const float*)d_in[1];
  const float* beta  = (const float*)d_in[2];
  const float* w_ih[3] = {(const float*)d_in[3], (const float*)d_in[7],  (const float*)d_in[11]};
  const float* w_hh[3] = {(const float*)d_in[4], (const float*)d_in[8],  (const float*)d_in[12]};
  const float* b_ih[3] = {(const float*)d_in[5], (const float*)d_in[9],  (const float*)d_in[13]};
  const float* b_hh[3] = {(const float*)d_in[6], (const float*)d_in[10], (const float*)d_in[14]};
  const float* fcw = (const float*)d_in[15];
  const float* fcb = (const float*)d_in[16];

  float* out = (float*)d_out;
  float* enc = out + (size_t)N_ * NC_;   // encoder_hidden region of d_out

  // ---- workspace carve (~235 MB) ----
  char* p = (char*)d_ws;
  auto carve = [&](size_t bytes) -> void* {
    void* r = (void*)p;
    p += (bytes + 255) & ~(size_t)255;
    return r;
  };
  float*    stats  = (float*)carve(2 * INP_ * sizeof(float));
  _Float16* xnorm  = (_Float16*)carve((size_t)N_ * T_ * INP_ * sizeof(_Float16));
  _Float16* actA   = (_Float16*)carve((size_t)N_ * T_ * C_ * sizeof(_Float16));
  _Float16* actB   = (_Float16*)carve((size_t)N_ * T_ * C_ * sizeof(_Float16));
  const int KinSrc[3] = {IN_, C_, C_};
  const int KinPad[3] = {INP_, C_, C_};
  _Float16* wih16[3];
  _Float16* whh16[3];
  for (int l = 0; l < 3; ++l) {
    wih16[l] = (_Float16*)carve((size_t)2 * 3 * H_ * KinPad[l] * sizeof(_Float16));
    whh16[l] = (_Float16*)carve((size_t)2 * 3 * H_ * H_ * sizeof(_Float16));
  }

  // ---- BN: stats then apply+pad+f16 ----
  bn_stats_kernel<<<IN_, 256, 0, stream>>>(x, gamma, beta, stats);
  bn_apply_kernel<<<(N_ * T_ * INP_ + 255) / 256, 256, 0, stream>>>(x, stats, xnorm);

  // ---- weight conversion to f16 (L2-resident working set ~92 MB) ----
  for (int l = 0; l < 3; ++l) {
    long nIh = (long)2 * 3 * H_ * KinPad[l];
    cvt_pad_kernel<<<(unsigned)((nIh + 255) / 256), 256, 0, stream>>>(
        w_ih[l], wih16[l], 2 * 3 * H_, KinSrc[l], KinPad[l]);
    long nHh = (long)2 * 3 * H_ * H_;
    cvt_pad_kernel<<<(unsigned)((nHh + 255) / 256), 256, 0, stream>>>(
        w_hh[l], whh16[l], 2 * 3 * H_, H_, H_);
  }

  // ---- 3 layers x 64 sequential steps (both directions per launch) ----
  const _Float16* layerIn[3]  = {xnorm, actA, actB};
  _Float16*       layerOut[3] = {actA, actB, actA};
  for (int l = 0; l < 3; ++l) {
    float* encPtr = (l == 2) ? enc : nullptr;
    for (int t = 0; t < T_; ++t) {
      gru_step_kernel<<<dim3(H_ / 16, N_ / 64, 2), 128, 0, stream>>>(
          layerIn[l], KinPad[l], layerOut[l], wih16[l], whh16[l],
          b_ih[l], b_hh[l], encPtr, t, t == 0 ? 1 : 0);
    }
  }

  // ---- FC head ----
  fc_kernel<<<N_, 64, 0, stream>>>(enc, fcw, fcb, out);
}
